// MambaLiteBlock_15169824489978
// MI455X (gfx1250) — compile-verified, matrix operand order checked by reading, not run on hardware
//
#include <hip/hip_runtime.h>
#include <hip/hip_bf16.h>

// ---------------------------------------------------------------------------
// MambaLite block for MI455X (gfx1250, wave32, WMMA).
// GEMMs: bf16 WMMA (v_wmma_f32_16x16x32_bf16), f32 accumulation.
// Block tile 128(M) x 256(N) x 32(K), 8 waves, wave tile 64x64 (16 WMMA/step).
// A tile staged with GLOBAL_LOAD_ASYNC_TO_LDS_B128 (ASYNCcnt), B tile staged
// through registers with K-pair interleave; double-buffered LDS.
// ---------------------------------------------------------------------------

typedef __attribute__((ext_vector_type(16))) __bf16 bf16x16;
typedef __attribute__((ext_vector_type(8)))  float  f32x8;
typedef __attribute__((ext_vector_type(8)))  unsigned int u32x8;
typedef int v4i32 __attribute__((vector_size(16)));

#define LDS_AS __attribute__((address_space(3)))
#define GLB_AS __attribute__((address_space(1)))

static constexpr int TBM = 128;   // block tile M
static constexpr int TBN = 256;   // block tile N
static constexpr int TBK = 32;    // block tile K (one WMMA K-step)

union FragCvt { u32x8 u; bf16x16 v; };

// ---------------------------------------------------------------------------
// Async global->LDS 16-byte copy + asynccnt wait
// ---------------------------------------------------------------------------
__device__ __forceinline__ void async_copy16(__bf16* lds_dst, const __bf16* gsrc) {
#if __has_builtin(__builtin_amdgcn_global_load_async_to_lds_b128)
  __builtin_amdgcn_global_load_async_to_lds_b128(
      (GLB_AS v4i32*)gsrc, (LDS_AS v4i32*)lds_dst, 0, 0);
#else
  unsigned loff = (unsigned)(unsigned long long)(LDS_AS void*)lds_dst;
  asm volatile("global_load_async_to_lds_b128 %0, %1, off"
               :: "v"(loff), "v"(gsrc) : "memory");
#endif
}

__device__ __forceinline__ void wait_async0() {
#if __has_builtin(__builtin_amdgcn_s_wait_asynccnt)
  __builtin_amdgcn_s_wait_asynccnt(0);
#else
  asm volatile("s_wait_asynccnt 0x0" ::: "memory");
#endif
}

// ---------------------------------------------------------------------------
// LayerNorm over D, emit bf16 (A operand of GEMM1)
// ---------------------------------------------------------------------------
__global__ __launch_bounds__(256) void ln_kernel(
    const float* __restrict__ x, const float* __restrict__ gamma,
    const float* __restrict__ beta, __bf16* __restrict__ xn, int D) {
  const int row = blockIdx.x;
  const float* xr = x + (size_t)row * D;
  __shared__ float red[256];

  float s = 0.f, ss = 0.f;
  for (int i = threadIdx.x; i < D; i += 256) {
    float v = xr[i];
    s += v;
    ss += v * v;
  }
  red[threadIdx.x] = s;
  __syncthreads();
  for (int off = 128; off > 0; off >>= 1) {
    if (threadIdx.x < off) red[threadIdx.x] += red[threadIdx.x + off];
    __syncthreads();
  }
  const float mean = red[0] / (float)D;
  __syncthreads();
  red[threadIdx.x] = ss;
  __syncthreads();
  for (int off = 128; off > 0; off >>= 1) {
    if (threadIdx.x < off) red[threadIdx.x] += red[threadIdx.x + off];
    __syncthreads();
  }
  const float var  = red[0] / (float)D - mean * mean;
  const float rstd = rsqrtf(var + 1e-5f);

  for (int i = threadIdx.x; i < D; i += 256) {
    float v = (xr[i] - mean) * rstd * gamma[i] + beta[i];
    xn[(size_t)row * D + i] = (__bf16)v;
  }
}

// ---------------------------------------------------------------------------
// f32 -> bf16 conversion (weights)
// ---------------------------------------------------------------------------
__global__ __launch_bounds__(256) void cvt_kernel(
    const float* __restrict__ src, __bf16* __restrict__ dst, size_t n) {
  size_t stride = (size_t)gridDim.x * blockDim.x;
  for (size_t i = (size_t)blockIdx.x * blockDim.x + threadIdx.x; i < n; i += stride)
    dst[i] = (__bf16)src[i];
}

// ---------------------------------------------------------------------------
// GEMM helpers
// ---------------------------------------------------------------------------

// A tile: TBM x TBK bf16 row-major; 16 contiguous bf16 per thread, staged via
// two async b128 global->LDS copies (no VGPR data path).
__device__ __forceinline__ void stage_A_async(const __bf16* __restrict__ A,
    int lda, int row0, int k0, int tid, __bf16* __restrict__ dstbase) {
  const int idx = tid * 16;
  const int rr = idx >> 5;         // TBK=32 elems per row
  const int cc = idx & 31;
  const __bf16* src = A + (size_t)(row0 + rr) * lda + (k0 + cc);
  async_copy16(dstbase + idx, src);
  async_copy16(dstbase + idx + 8, src + 8);
}

// B tile: TBK x TBN bf16 row-major in global; staged pair-interleaved n-major:
//   sB32[n * 16 + p] = B[k0+2p][n] | (B[k0+2p+1][n] << 16),  p = 0..15
// so a lane's fragment (K-pairs for one n) is 2 aligned b128 LDS loads (uint4 h and 2+h).
__device__ __forceinline__ void stage_B_load(const __bf16* __restrict__ B,
    int ldb, int col0, int k0, int tid, uint4 r[2][2]) {
#pragma unroll
  for (int u = 0; u < 2; ++u) {
    const int unit = tid + u * 256;          // 512 (p, n-group) units
    const int p = unit & 15;                 // K-pair index
    const int n = (unit >> 4) * 8;           // 8-column group
    const uint4* r0 =
        reinterpret_cast<const uint4*>(B + (size_t)(k0 + 2 * p) * ldb + (col0 + n));
    const uint4* r1 =
        reinterpret_cast<const uint4*>(B + (size_t)(k0 + 2 * p + 1) * ldb + (col0 + n));
    r[u][0] = r0[0];
    r[u][1] = r1[0];
  }
}
__device__ __forceinline__ void stage_B_store(unsigned int* __restrict__ sB32,
                                              int tid, const uint4 r[2][2]) {
#pragma unroll
  for (int u = 0; u < 2; ++u) {
    const int unit = tid + u * 256;
    const int p = unit & 15;
    const int n = (unit >> 4) * 8;
    const unsigned int* a = reinterpret_cast<const unsigned int*>(&r[u][0]);
    const unsigned int* b = reinterpret_cast<const unsigned int*>(&r[u][1]);
#pragma unroll
    for (int i = 0; i < 4; ++i) {
      const unsigned int lo = (a[i] & 0xffffu) | (b[i] << 16);
      const unsigned int hi = (a[i] >> 16) | (b[i] & 0xffff0000u);
      sB32[(n + 2 * i) * 16 + p]     = lo;
      sB32[(n + 2 * i + 1) * 16 + p] = hi;
    }
  }
}

// Fragment load: 8 K-pairs per lane, pair index pk = ((r>>2)<<3)+(h<<2)+(r&3)
// -> two aligned b128 LDS loads at uint4 indices h and 2+h.
__device__ __forceinline__ bf16x16 load_frag(const void* base, int h) {
  const uint4* p = reinterpret_cast<const uint4*>(base);
  const uint4 lo = p[h];
  const uint4 hi = p[2 + h];
  FragCvt f;
  f.u[0] = lo.x; f.u[1] = lo.y; f.u[2] = lo.z; f.u[3] = lo.w;
  f.u[4] = hi.x; f.u[5] = hi.y; f.u[6] = hi.z; f.u[7] = hi.w;
  return f.v;
}

// ---------------------------------------------------------------------------
// Tiled WMMA GEMM: C[M,N] = A[M,K](bf16) * B[K,N](bf16) + bias[N] (+ residual)
// 256 threads (8 waves, 2x4), wave tile 64x64 = 4x4 WMMA tiles.
// ---------------------------------------------------------------------------
__global__ __launch_bounds__(256, 1) void gemm_bf16_kernel(
    const __bf16* __restrict__ A, const __bf16* __restrict__ B,
    const float* __restrict__ bias, const float* __restrict__ residual,
    float* __restrict__ C, int M, int N, int K, int lda, int ldb, int ldc) {
  __shared__ __align__(16) __bf16 sA[2][TBM * TBK];          // 2 x 8 KB
  __shared__ __align__(16) unsigned int sB32[2][TBN * 16];   // 2 x 16 KB

  const int row0 = blockIdx.y * TBM;
  const int col0 = blockIdx.x * TBN;
  const int tid  = threadIdx.x;
  const int lane = tid & 31;
  const int wave = tid >> 5;
  const int m_wave = (wave & 1) * 64;        // 2 waves along M
  const int n_wave = (wave >> 1) * 64;       // 4 waves along N
  const int lm = lane & 15;                  // M/N index within 16x16 tile
  const int h  = lane >> 4;                  // K-half selector

  f32x8 acc[4][4];
#pragma unroll
  for (int i = 0; i < 4; ++i)
#pragma unroll
    for (int j = 0; j < 4; ++j)
#pragma unroll
      for (int e = 0; e < 8; ++e) acc[i][j][e] = 0.f;

  uint4 rB[2][2];
  stage_A_async(A, lda, row0, 0, tid, sA[0]);
  stage_B_load(B, ldb, col0, 0, tid, rB);
  stage_B_store(sB32[0], tid, rB);

  int cur = 0;
  for (int k0 = 0; k0 < K; k0 += TBK) {
    wait_async0();     // own async writes into sA[cur] complete
    __syncthreads();   // everyone's staging of buffer [cur] visible

    const bool has_next = (k0 + TBK) < K;
    if (has_next) {
      // all reads of buffer [cur^1] finished before the barrier above
      stage_A_async(A, lda, row0, k0 + TBK, tid, sA[cur ^ 1]);
      stage_B_load(B, ldb, col0, k0 + TBK, tid, rB);
    }
    if (k0 + 2 * TBK < K) {    // pull the tile after next into cache
      __builtin_prefetch(A + (size_t)(row0 + (tid >> 1)) * lda + (k0 + 2 * TBK), 0, 0);
      __builtin_prefetch(B + (size_t)(k0 + 2 * TBK + (tid >> 5)) * ldb + col0 + ((tid & 31) << 3), 0, 0);
    }

    bf16x16 af[4], bfr[4];
#pragma unroll
    for (int i = 0; i < 4; ++i)
      af[i] = load_frag(&sA[cur][(m_wave + i * 16 + lm) * TBK], h);
#pragma unroll
    for (int j = 0; j < 4; ++j)
      bfr[j] = load_frag(&sB32[cur][(n_wave + j * 16 + lm) * 16], h);

#pragma unroll
    for (int i = 0; i < 4; ++i)
#pragma unroll
      for (int j = 0; j < 4; ++j)
        acc[i][j] = __builtin_amdgcn_wmma_f32_16x16x32_bf16(
            false, af[i], false, bfr[j], (short)0, acc[i][j], false, false);

    if (has_next)              // write other buffer; readers sync at loop top
      stage_B_store(sB32[cur ^ 1], tid, rB);
    cur ^= 1;
  }

  // epilogue: C/D layout: VGPR r -> M = r + 8*h, N = lane%16
#pragma unroll
  for (int i = 0; i < 4; ++i)
#pragma unroll
    for (int j = 0; j < 4; ++j)
#pragma unroll
      for (int r = 0; r < 8; ++r) {
        const int mm = row0 + m_wave + i * 16 + r + 8 * h;
        const int nn = col0 + n_wave + j * 16 + lm;
        float v = acc[i][j][r] + bias[nn];
        if (residual) v += residual[(size_t)mm * ldc + nn];
        C[(size_t)mm * ldc + nn] = v;
      }
}

// ---------------------------------------------------------------------------
// Causal depthwise conv K=7 over T. u holds [g|v] (U = 2H); v at offset H.
// ---------------------------------------------------------------------------
__global__ __launch_bounds__(256) void dwconv_kernel(
    const float* __restrict__ u, const float* __restrict__ w,
    const float* __restrict__ b, float* __restrict__ out,
    int BT, int T, int H, int U) {
  const size_t total = (size_t)BT * H;
  const size_t stride = (size_t)gridDim.x * blockDim.x;
  for (size_t i = (size_t)blockIdx.x * blockDim.x + threadIdx.x; i < total; i += stride) {
    const int hh  = (int)(i % H);
    const int row = (int)(i / H);     // b*T + t
    const int t   = row % T;
    float acc = b[hh];
#pragma unroll
    for (int k = 0; k < 7; ++k) {
      const int tt = t - 6 + k;
      if (tt >= 0)
        acc += u[(size_t)(row - 6 + k) * U + H + hh] * w[hh * 7 + k];
    }
    out[i] = acc;
  }
}

// ---------------------------------------------------------------------------
// Per-channel decay scan: y_t = a*y_{t-1} + v_t ; emit bf16 (A of GEMM2)
// ---------------------------------------------------------------------------
__global__ __launch_bounds__(256) void scan_kernel(
    const float* __restrict__ u, const float* __restrict__ log_decay,
    __bf16* __restrict__ y, int Bc, int T, int H, int U) {
  const int i = blockIdx.x * blockDim.x + threadIdx.x;
  if (i >= Bc * H) return;
  const int hh = i % H;
  const int b  = i / H;
  const float a = 1.f / (1.f + __expf(-log_decay[hh]));
  const float* vp = u + (size_t)b * T * U + H + hh;
  __bf16* yp = y + (size_t)b * T * H + hh;
  float st = 0.f;
  for (int t = 0; t < T; ++t) {
    st = st * a + vp[(size_t)t * U];
    yp[(size_t)t * H] = (__bf16)st;
  }
}

// ---------------------------------------------------------------------------
// y2 = sigmoid(g) * (conv + ssm) -> bf16 (A of GEMM3)
// ---------------------------------------------------------------------------
__global__ __launch_bounds__(256) void combine_kernel(
    const float* __restrict__ u, const float* __restrict__ conv,
    const float* __restrict__ s, __bf16* __restrict__ y2,
    size_t total, int H, int U) {
  const size_t stride = (size_t)gridDim.x * blockDim.x;
  for (size_t i = (size_t)blockIdx.x * blockDim.x + threadIdx.x; i < total; i += stride) {
    const int hh = (int)(i % H);
    const size_t row = i / H;
    const float g = 1.f / (1.f + __expf(-u[row * U + hh]));
    y2[i] = (__bf16)(g * (conv[i] + s[i]));
  }
}

// ---------------------------------------------------------------------------
extern "C" void kernel_launch(void* const* d_in, const int* in_sizes, int n_in,
                              void* d_out, int out_size, void* d_ws, size_t ws_size,
                              hipStream_t stream) {
  const float* x         = (const float*)d_in[0];
  const float* gamma     = (const float*)d_in[1];
  const float* beta      = (const float*)d_in[2];
  const float* in_w      = (const float*)d_in[3];
  const float* in_b      = (const float*)d_in[4];
  const float* dw_w      = (const float*)d_in[5];
  const float* dw_b      = (const float*)d_in[6];
  const float* log_decay = (const float*)d_in[7];
  const float* mix_w     = (const float*)d_in[8];
  const float* mix_b     = (const float*)d_in[9];
  const float* out_w     = (const float*)d_in[10];
  const float* out_b     = (const float*)d_in[11];
  float* out = (float*)d_out;

  const int Bc = 4, T = 2048, D = 1024, H = 2048, U = 4096;
  const int BT = Bc * T;   // 8192

  char* ws = (char*)d_ws;
  size_t off = 0;
  auto take = [&](size_t bytes) -> char* {
    char* p = ws + off;
    off += (bytes + 255) & ~(size_t)255;
    return p;
  };
  __bf16* xnb   = (__bf16*)take((size_t)BT * D * 2);   // 16 MB
  __bf16* inwb  = (__bf16*)take((size_t)D * U * 2);    //  8 MB
  __bf16* mixwb = (__bf16*)take((size_t)H * H * 2);    //  8 MB
  __bf16* outwb = (__bf16*)take((size_t)H * D * 2);    //  4 MB
  float*  u     = (float*)take((size_t)BT * U * 4);    // 128 MB
  float*  conv  = (float*)take((size_t)BT * H * 4);    // 64 MB
  __bf16* yscan = (__bf16*)take((size_t)BT * H * 2);   // 32 MB
  float*  sbuf  = (float*)take((size_t)BT * H * 4);    // 64 MB
  __bf16* y2    = (__bf16*)take((size_t)BT * H * 2);   // 32 MB

  // 1. LayerNorm -> bf16 activations
  ln_kernel<<<BT, 256, 0, stream>>>(x, gamma, beta, xnb, D);

  // 2. Weight conversions to bf16
  cvt_kernel<<<2048, 256, 0, stream>>>(in_w,  inwb,  (size_t)D * U);
  cvt_kernel<<<2048, 256, 0, stream>>>(mix_w, mixwb, (size_t)H * H);
  cvt_kernel<<<1024, 256, 0, stream>>>(out_w, outwb, (size_t)H * D);

  // 3. GEMM1: u = xn @ in_w + in_b          (8192 x 4096 x 1024)
  dim3 g1(U / TBN, BT / TBM);
  gemm_bf16_kernel<<<g1, 256, 0, stream>>>(xnb, inwb, in_b, nullptr, u,
                                           BT, U, D, D, U, U);

  // 4. Depthwise causal conv on v
  dwconv_kernel<<<4096, 256, 0, stream>>>(u, dw_w, dw_b, conv, BT, T, H, U);

  // 5. Decay scan on v -> bf16
  scan_kernel<<<(Bc * H) / 256, 256, 0, stream>>>(u, log_decay, yscan, Bc, T, H, U);

  // 6. GEMM2: s = scan @ mix_w + mix_b      (8192 x 2048 x 2048)
  dim3 g2(H / TBN, BT / TBM);
  gemm_bf16_kernel<<<g2, 256, 0, stream>>>(yscan, mixwb, mix_b, nullptr, sbuf,
                                           BT, H, H, H, H, H);

  // 7. y2 = sigmoid(g) * (conv + s) -> bf16
  combine_kernel<<<4096, 256, 0, stream>>>(u, conv, sbuf, y2, (size_t)BT * H, H, U);

  // 8. GEMM3: out = y2 @ out_w + out_b + x  (8192 x 1024 x 2048)
  dim3 g3(D / TBN, BT / TBM);
  gemm_bf16_kernel<<<g3, 256, 0, stream>>>(y2, outwb, out_b, x, out,
                                           BT, D, H, H, D, D);
}
